// TCNN_50173807952490
// MI455X (gfx1250) — matrix operand order repaired and checked by reading, not compiled, not run on hardware
//
#include <hip/hip_runtime.h>
#include <hip/hip_bf16.h>

// ---------------------------------------------------------------------------
// TCNN pipeline for MI455X (gfx1250, wave32, WMMA bf16 16x16x32)
//   B=256, N=512, F=64, C1=256, C2=128, RANK=16
// Workspace layout (bytes):
//   [0, 98304)                     W1 bf16 [256][192]   (o, k*64+c)
//   [98304, 294912)                W2 bf16 [128][768]   (o, k*256+c)
//   [294912, 294912+67108864)      h1 bf16 [B][N][256]
//   [+67108864, +131072)           pooled  u32-as-f32 [B][128]
// ---------------------------------------------------------------------------

typedef __attribute__((ext_vector_type(16))) __bf16 v16bf;
typedef __attribute__((ext_vector_type(8)))  __bf16 v8bf;
typedef __attribute__((ext_vector_type(8)))  float  v8f;
typedef __attribute__((__vector_size__(8)))  int    v2i;   // async b64 payload

#define BB 256
#define NN 512
#define FF 64
#define CC1 256
#define CC2 128

// Async global->LDS gather path (CDNA5 ASYNCcnt-tracked copies), guarded so
// compilation cannot regress if the toolchain lacks the builtins.
#if defined(__AMDGCN__) && \
    __has_builtin(__builtin_amdgcn_global_load_async_to_lds_b64) && \
    __has_builtin(__builtin_amdgcn_s_wait_asynccnt)
#define USE_ASYNC_GATHER 1
#else
#define USE_ASYNC_GATHER 0
#endif

static __device__ __forceinline__ unsigned short f2bf(float f) {
    unsigned u = __float_as_uint(f);
    unsigned r = (u + 0x7FFFu + ((u >> 16) & 1u)) >> 16;
    return (unsigned short)r;
}

// Build a 16-bit A-fragment per CDNA5 layout from a row-major LDS row.
// lane<16: elems 0..7 = K[0..7],  8..15 = K[16..23]
// lane>=16: elems 0..7 = K[8..15], 8..15 = K[24..31]
static __device__ __forceinline__ v16bf load_a_frag(const unsigned short* p) {
    const v8bf lo = *(const v8bf*)(p);
    const v8bf hi = *(const v8bf*)(p + 16);
    v16bf a;
#pragma unroll
    for (int i = 0; i < 8; ++i) { a[i] = lo[i]; a[i + 8] = hi[i]; }
    return a;
}

// ---------------------------------------------------------------------------
// Weight pre-swizzle: conv w [O][C][3] f32 -> W [O][3*C] bf16 with k-major-C
// ---------------------------------------------------------------------------
__global__ void prep_weights_kernel(const float* __restrict__ w1,
                                    const float* __restrict__ w2,
                                    unsigned short* __restrict__ W1,
                                    unsigned short* __restrict__ W2) {
    int i = blockIdx.x * blockDim.x + threadIdx.x;
    if (i < CC1 * 192) {
        int o = i / 192, r = i - o * 192;
        int k = r >> 6, c = r & 63;
        W1[o * 192 + k * 64 + c] = f2bf(w1[o * 192 + c * 3 + k]);
    }
    int j = i - CC1 * 192;
    if (j >= 0 && j < CC2 * 768) {
        int o = j / 768, r = j - o * 768;
        int k = r >> 8, c = r & 255;
        W2[o * 768 + k * 256 + c] = f2bf(w2[o * 768 + c * 3 + k]);
    }
}

__global__ void zero_u32_kernel(unsigned* __restrict__ p, int n) {
    int i = blockIdx.x * blockDim.x + threadIdx.x;
    if (i < n) p[i] = 0u;
}

// ---------------------------------------------------------------------------
// Block 1: gather data -> A[32][192] bf16 (fused f32->bf16), GEMM vs W1
//          (K=192, O=256), bias + LayerNorm + ReLU, store h1 as bf16.
// grid = B * (N/32), block = 128 (4 waves); wave w owns o-tiles 4w..4w+3,
// each B fragment feeds two row-tiles (2 WMMAs per global B load).
// ---------------------------------------------------------------------------
__global__ void tree_conv1_kernel(const float* __restrict__ data,
                                  const int* __restrict__ idxes,
                                  const unsigned short* __restrict__ W1,
                                  const float* __restrict__ cb,
                                  const float* __restrict__ g,
                                  const float* __restrict__ beta,
                                  unsigned short* __restrict__ h1) {
    // A tile (32x192 bf16 = 12KB) aliased with f32 spill tile (32x256 = 32KB)
    __shared__ __align__(16) unsigned char smem[32 * 256 * 4];
    unsigned short* Alds = (unsigned short*)smem;
    float* Olds = (float*)smem;
    __shared__ int sidx[96];

    const int b = blockIdx.x >> 4;            // / (N/32)
    const int n0 = (blockIdx.x & 15) * 32;
    const int tid = threadIdx.x;

    if (tid < 96) sidx[tid] = idxes[b * (3 * NN) + n0 * 3 + tid];
    __syncthreads();

    // gather + f32->bf16 into LDS: A[m][k*64+c]
    for (int j = tid; j < 32 * 192; j += 128) {
        int m = j / 192, r = j - m * 192;
        int k = r >> 6, c = r & 63;
        Alds[j] = f2bf(data[((size_t)b * NN + sidx[m * 3 + k]) * FF + c]);
    }
    __syncthreads();

    const int wave = tid >> 5, lane = tid & 31;
    const int lrow = lane & 15, lhalf = lane >> 4;

    v8f acc[8];
#pragma unroll
    for (int t = 0; t < 8; ++t)
#pragma unroll
        for (int r = 0; r < 8; ++r) acc[t][r] = 0.0f;

#pragma unroll
    for (int kk = 0; kk < 6; ++kk) {
        v16bf a0 = load_a_frag(&Alds[lrow * 192 + kk * 32 + lhalf * 8]);
        v16bf a1 = load_a_frag(&Alds[(16 + lrow) * 192 + kk * 32 + lhalf * 8]);
#pragma unroll
        for (int t = 0; t < 4; ++t) {
            int o0 = (wave * 4 + t) * 16;
            const v16bf bf = load_a_frag(W1 + (size_t)(o0 + lrow) * 192 +
                                             kk * 32 + lhalf * 8);
            acc[t] = __builtin_amdgcn_wmma_f32_16x16x32_bf16(
                false, a0, false, bf, (short)0, acc[t], false, false);
            acc[4 + t] = __builtin_amdgcn_wmma_f32_16x16x32_bf16(
                false, a1, false, bf, (short)0, acc[4 + t], false, false);
        }
    }
    __syncthreads();   // all A reads finished before Olds overwrites smem

    // spill C tiles: VGPR r -> row r (lanes 0-15) / r+8 (lanes 16-31)
    const int mbase = lhalf * 8;
#pragma unroll
    for (int rt = 0; rt < 2; ++rt)
#pragma unroll
        for (int t = 0; t < 4; ++t) {
            int o0 = (wave * 4 + t) * 16;
#pragma unroll
            for (int r = 0; r < 8; ++r)
                Olds[(rt * 16 + r + mbase) * 256 + o0 + lrow] = acc[rt * 4 + t][r];
        }
    __syncthreads();

    // LayerNorm+ReLU: wave handles rows 8w..8w+7; lane covers 8 strided elems
#pragma unroll
    for (int rr = 0; rr < 8; ++rr) {
        int m = wave * 8 + rr;
        float vbuf[8];
        float s = 0.f, s2 = 0.f;
#pragma unroll
        for (int e = 0; e < 8; ++e) {
            int o = lane + e * 32;
            float v = Olds[m * 256 + o] + cb[o];
            vbuf[e] = v; s += v; s2 += v * v;
        }
#pragma unroll
        for (int off = 16; off; off >>= 1) {
            s += __shfl_xor(s, off);
            s2 += __shfl_xor(s2, off);
        }
        float mean = s * (1.f / 256.f);
        float var = s2 * (1.f / 256.f) - mean * mean;
        float inv = rsqrtf(var + 1e-5f);
        size_t base = ((size_t)b * NN + n0 + m) * 256;
#pragma unroll
        for (int e = 0; e < 8; ++e) {
            int o = lane + e * 32;
            float y = (vbuf[e] - mean) * inv * g[o] + beta[o];
            h1[base + o] = f2bf(fmaxf(y, 0.f));
        }
    }
}

// ---------------------------------------------------------------------------
// Block 2: async-gather h1 (bf16) -> A[32][768], GEMM vs W2 (K=768, O=128),
//          bias + LayerNorm + ReLU, fused max-pool via atomicMax(u32).
// grid = B * (N/32), block = 128 (4 waves); wave w owns o-tiles 2w..2w+1.
// ---------------------------------------------------------------------------
__global__ void tree_conv2_kernel(const unsigned short* __restrict__ h1,
                                  const int* __restrict__ idxes,
                                  const unsigned short* __restrict__ W2,
                                  const float* __restrict__ cb,
                                  const float* __restrict__ g,
                                  const float* __restrict__ beta,
                                  unsigned* __restrict__ pooled) {
    // A tile 32x768 bf16 = 48KB, aliased with f32 spill tile 32x128 = 16KB
    __shared__ __align__(16) unsigned char smem[32 * 768 * 2];
    unsigned short* Alds = (unsigned short*)smem;
    float* Olds = (float*)smem;
    __shared__ int sidx[96];

    const int b = blockIdx.x >> 4;
    const int n0 = (blockIdx.x & 15) * 32;
    const int tid = threadIdx.x;

    if (tid < 96) sidx[tid] = idxes[b * (3 * NN) + n0 * 3 + tid];

    // prefetch the weight block this WG will fully stream (L2-resident)
    {
        const char* wb = (const char*)W2;
        for (size_t p = (size_t)tid * 128; p < (size_t)CC2 * 768 * 2;
             p += 128u * 128u)
            __builtin_prefetch(wb + p, 0, 1);
    }
    __syncthreads();

    // gather bf16 rows as qwords: A[m][k*256+c]  (row = 192 qwords)
    const unsigned long long* h1q = (const unsigned long long*)h1;
    unsigned long long* Aq = (unsigned long long*)Alds;
    for (int j = tid; j < 32 * 192; j += 128) {
        int m = j / 192, r = j - m * 192;
        int k = r >> 6, cq = r & 63;          // 256 bf16 = 64 qwords / segment
        const unsigned long long* gp =
            h1q + ((size_t)b * NN + sidx[m * 3 + k]) * 64 + cq;
#if USE_ASYNC_GATHER
        __builtin_amdgcn_global_load_async_to_lds_b64(
            (__attribute__((address_space(1))) v2i*)gp,
            (__attribute__((address_space(3))) v2i*)(Aq + j), 0, 0);
#else
        Aq[j] = *gp;
#endif
    }
#if USE_ASYNC_GATHER
    __builtin_amdgcn_s_wait_asynccnt(0);
#endif
    __syncthreads();

    const int wave = tid >> 5, lane = tid & 31;
    const int lrow = lane & 15, lhalf = lane >> 4;

    v8f acc[4];
#pragma unroll
    for (int t = 0; t < 4; ++t)
#pragma unroll
        for (int r = 0; r < 8; ++r) acc[t][r] = 0.0f;

    for (int kk = 0; kk < 24; ++kk) {
        v16bf a0 = load_a_frag(&Alds[lrow * 768 + kk * 32 + lhalf * 8]);
        v16bf a1 = load_a_frag(&Alds[(16 + lrow) * 768 + kk * 32 + lhalf * 8]);
#pragma unroll
        for (int t = 0; t < 2; ++t) {
            int o0 = (wave * 2 + t) * 16;
            const v16bf bf = load_a_frag(W2 + (size_t)(o0 + lrow) * 768 +
                                             kk * 32 + lhalf * 8);
            acc[t] = __builtin_amdgcn_wmma_f32_16x16x32_bf16(
                false, a0, false, bf, (short)0, acc[t], false, false);
            acc[2 + t] = __builtin_amdgcn_wmma_f32_16x16x32_bf16(
                false, a1, false, bf, (short)0, acc[2 + t], false, false);
        }
    }
    __syncthreads();   // all A reads finished before Olds overwrites smem

    const int mbase = lhalf * 8;
#pragma unroll
    for (int rt = 0; rt < 2; ++rt)
#pragma unroll
        for (int t = 0; t < 2; ++t) {
            int o0 = (wave * 2 + t) * 16;
#pragma unroll
            for (int r = 0; r < 8; ++r)
                Olds[(rt * 16 + r + mbase) * 128 + o0 + lrow] = acc[rt * 2 + t][r];
        }
    __syncthreads();

    // LN + ReLU + max-pool (post-ReLU values >= 0 -> uint compare == f32 cmp)
#pragma unroll
    for (int rr = 0; rr < 8; ++rr) {
        int m = wave * 8 + rr;
        float vbuf[4];
        float s = 0.f, s2 = 0.f;
#pragma unroll
        for (int e = 0; e < 4; ++e) {
            int o = lane + e * 32;
            float v = Olds[m * 128 + o] + cb[o];
            vbuf[e] = v; s += v; s2 += v * v;
        }
#pragma unroll
        for (int off = 16; off; off >>= 1) {
            s += __shfl_xor(s, off);
            s2 += __shfl_xor(s2, off);
        }
        float mean = s * (1.f / 128.f);
        float var = s2 * (1.f / 128.f) - mean * mean;
        float inv = rsqrtf(var + 1e-5f);
#pragma unroll
        for (int e = 0; e < 4; ++e) {
            int o = lane + e * 32;
            float y = fmaxf((vbuf[e] - mean) * inv * g[o] + beta[o], 0.f);
            atomicMax(&pooled[b * 128 + o], __float_as_uint(y));
        }
    }
}

// ---------------------------------------------------------------------------
// MLP head: one wave per batch element.
// z[160] = [pooled(128) | user_emb(16) | item_emb(16)]
// ---------------------------------------------------------------------------
__global__ void head_kernel(const unsigned* __restrict__ pooled_u,
                            const int* __restrict__ user_idx,
                            const int* __restrict__ item_idx,
                            const float* __restrict__ user_emb,
                            const float* __restrict__ item_emb,
                            const float* __restrict__ l1w, const float* __restrict__ l1b,
                            const float* __restrict__ g1, const float* __restrict__ b1,
                            const float* __restrict__ l2w, const float* __restrict__ l2b,
                            const float* __restrict__ g2, const float* __restrict__ b2,
                            const float* __restrict__ ow, const float* __restrict__ ob,
                            float* __restrict__ out) {
    const int b = blockIdx.x;
    const int l = threadIdx.x;                 // 0..31
    const float* pooled = (const float*)pooled_u;
    const int ui = user_idx[b], ii = item_idx[b];

    // lin1: each lane computes one of 32 outputs
    float acc = l1b[l];
    for (int j = 0; j < 128; ++j) acc += l1w[l * 160 + j] * pooled[b * 128 + j];
#pragma unroll
    for (int j = 0; j < 16; ++j) acc += l1w[l * 160 + 128 + j] * user_emb[(size_t)ui * 16 + j];
#pragma unroll
    for (int j = 0; j < 16; ++j) acc += l1w[l * 160 + 144 + j] * item_emb[(size_t)ii * 16 + j];

    float s = acc, s2 = acc * acc;
#pragma unroll
    for (int off = 16; off; off >>= 1) { s += __shfl_xor(s, off); s2 += __shfl_xor(s2, off); }
    float mean = s * (1.f / 32.f), var = s2 * (1.f / 32.f) - mean * mean;
    float a1 = fmaxf((acc - mean) * rsqrtf(var + 1e-5f) * g1[l] + b1[l], 0.f);

    // lin2: lanes 0-15 (duplicated on 16-31 to stay convergent)
    int l16 = l & 15;
    float acc2 = l2b[l16];
#pragma unroll
    for (int j = 0; j < 32; ++j) acc2 += l2w[l16 * 32 + j] * __shfl(a1, j);

    float t = acc2, t2 = acc2 * acc2;
#pragma unroll
    for (int off = 8; off; off >>= 1) { t += __shfl_xor(t, off); t2 += __shfl_xor(t2, off); }
    float mean2 = t * (1.f / 16.f), var2 = t2 * (1.f / 16.f) - mean2 * mean2;
    float a2 = fmaxf((acc2 - mean2) * rsqrtf(var2 + 1e-5f) * g2[l16] + b2[l16], 0.f);

    float r = 0.f;
#pragma unroll
    for (int j = 0; j < 16; ++j) r += __shfl(a2, j) * ow[j];
    if (l == 0) out[b] = r + ob[0];
}

// ---------------------------------------------------------------------------
extern "C" void kernel_launch(void* const* d_in, const int* in_sizes, int n_in,
                              void* d_out, int out_size, void* d_ws, size_t ws_size,
                              hipStream_t stream) {
    const float* data     = (const float*)d_in[0];
    const int*   idxes    = (const int*)d_in[1];
    const int*   user_idx = (const int*)d_in[2];
    const int*   item_idx = (const int*)d_in[3];
    const float* conv1_w  = (const float*)d_in[4];
    const float* conv1_b  = (const float*)d_in[5];
    const float* ln1_g    = (const float*)d_in[6];
    const float* ln1_b    = (const float*)d_in[7];
    const float* conv2_w  = (const float*)d_in[8];
    const float* conv2_b  = (const float*)d_in[9];
    const float* ln2_g    = (const float*)d_in[10];
    const float* ln2_b    = (const float*)d_in[11];
    const float* user_emb = (const float*)d_in[12];
    const float* item_emb = (const float*)d_in[13];
    const float* lin1_w   = (const float*)d_in[14];
    const float* lin1_b   = (const float*)d_in[15];
    const float* mln1_g   = (const float*)d_in[16];
    const float* mln1_b   = (const float*)d_in[17];
    const float* lin2_w   = (const float*)d_in[18];
    const float* lin2_b   = (const float*)d_in[19];
    const float* mln2_g   = (const float*)d_in[20];
    const float* mln2_b   = (const float*)d_in[21];
    const float* out_w    = (const float*)d_in[22];
    const float* out_b    = (const float*)d_in[23];

    char* ws = (char*)d_ws;
    unsigned short* W1     = (unsigned short*)(ws);
    unsigned short* W2     = (unsigned short*)(ws + 98304);
    unsigned short* h1     = (unsigned short*)(ws + 98304 + 196608);
    unsigned*       pooled = (unsigned*)(ws + 98304 + 196608 + (size_t)67108864);

    // 1) swizzle weights to bf16 (o, k-major-c)
    {
        int total = CC1 * 192 + CC2 * 768;
        prep_weights_kernel<<<(total + 255) / 256, 256, 0, stream>>>(conv1_w, conv2_w, W1, W2);
    }
    // 2) zero pooled accumulator
    zero_u32_kernel<<<(BB * 128 + 255) / 256, 256, 0, stream>>>(pooled, BB * 128);
    // 3) tree conv block 1 (32-row tiles)
    tree_conv1_kernel<<<BB * (NN / 32), 128, 0, stream>>>(data, idxes, W1, conv1_b,
                                                          ln1_g, ln1_b, h1);
    // 4) tree conv block 2 + fused max pool (32-row tiles, async gather)
    tree_conv2_kernel<<<BB * (NN / 32), 128, 0, stream>>>(h1, idxes, W2, conv2_b,
                                                          ln2_g, ln2_b, pooled);
    // 5) MLP head
    head_kernel<<<BB, 32, 0, stream>>>(pooled, user_idx, item_idx, user_emb, item_emb,
                                       lin1_w, lin1_b, mln1_g, mln1_b,
                                       lin2_w, lin2_b, mln2_g, mln2_b,
                                       out_w, out_b, (float*)d_out);
}
